// TrainTest_42159398978097
// MI455X (gfx1250) — matrix operand order repaired and checked by reading, not compiled, not run on hardware
//
#include <hip/hip_runtime.h>

// ---------------- problem constants ----------------
#define EMBED 1024
#define HID   16000
#define BB    16
#define TT    256
#define STEPS (TT - 1)        // 255
#define ROWS  (STEPS * BB)    // 4080
#define NTILES (HID / 16)     // 1000
#define NGRP  (NTILES / 4)    // 250 groups of 4 tiles

typedef __attribute__((ext_vector_type(16))) __bf16 v16bf;
typedef __attribute__((ext_vector_type(8)))  float  v8f;

// ---------------- WMMA helpers (CDNA5 layouts, wave32) ----------------
// A fragment: 16x32 bf16, row-major source with leading dim ldk (in elements).
// lanes 0-15: M=lane,  K = k0 + {0..7, 16..23}
// lanes 16-31: M=lane-16, K = k0 + {8..15, 24..31}
__device__ inline v16bf load_a_frag(const __bf16* A, int ldk, int k0) {
  int lane = threadIdx.x & 31;
  int m = lane & 15;
  const __bf16* p = A + m * ldk + k0 + ((lane >> 4) << 3);
  union { v16bf v; uint4 q[2]; } u;
  u.q[0] = *(const uint4*)p;
  u.q[1] = *(const uint4*)(p + 16);
  return u.v;
}

// B fragment: 32x16 bf16, row-major source (K rows of N) with leading dim ldn.
// N = n0 + lane%16; K = k0 + {0..7, 16..23} (lanes 0-15), k0 + {8..15, 24..31} (lanes 16-31).
__device__ inline v16bf load_b_frag(const __bf16* Bm, long ldn, int k0, int n0) {
  int lane = threadIdx.x & 31;
  const __bf16* p = Bm + (long)(k0 + ((lane >> 4) << 3)) * ldn + n0 + (lane & 15);
  v16bf v;
#pragma unroll
  for (int s = 0; s < 16; ++s) v[s] = p[(long)((s & 7) + ((s >> 3) << 4)) * ldn];
  return v;
}

__device__ inline v8f wmma_bf16(v16bf a, v16bf b, v8f c) {
  return __builtin_amdgcn_wmma_f32_16x16x32_bf16(false, a, false, b, (short)0, c,
                                                 false, false);
}

// 4 N-tile group GEMM: one A-frag load feeds 4 WMMAs (4x LDS traffic reduction)
__device__ inline void gemm_grp4(const __bf16* sA, const __bf16* Vb, int n0,
                                 v8f acc[4]) {
  for (int kk = 0; kk < EMBED; kk += 32) {
    if (kk + 32 < EMBED)
      __builtin_prefetch((const void*)(Vb + (long)(kk + 32 + (threadIdx.x & 31)) * HID + n0), 0, 3);
    v16bf a = load_a_frag(sA, EMBED, kk);
#pragma unroll
    for (int q = 0; q < 4; ++q) {
      v16bf b = load_b_frag(Vb, HID, kk, n0 + 16 * q);
      acc[q] = wmma_bf16(a, b, acc[q]);
    }
  }
}

// monotone float<->signed-int key for atomicMax-based float max
__device__ inline int fkey(float f) {
  int i = __float_as_int(f);
  return (i < 0) ? (i ^ 0x7FFFFFFF) : i;
}
__device__ inline float funkey(int k) {
  int i = (k < 0) ? (k ^ 0x7FFFFFFF) : k;
  return __int_as_float(i);
}

// ---------------- kernels ----------------
__global__ void k_zero(float* out) {
  if (threadIdx.x < 2) out[threadIdx.x] = 0.0f;
}

__global__ void k_convert(const float* __restrict__ src, __bf16* __restrict__ dst, int n) {
  for (int i = blockIdx.x * blockDim.x + threadIdx.x; i < n; i += gridDim.x * blockDim.x)
    dst[i] = (__bf16)src[i];
}

// gather embedding rows: Emb[(t*16+b), :] = X[data[b][t], :]
__global__ void k_embed(const int* __restrict__ data, const float* __restrict__ X,
                        __bf16* __restrict__ Emb) {
  int row = blockIdx.x;             // 0..4079
  int t = row >> 4, b = row & 15;
  int tok = data[b * TT + t];
  const float* src = X + (long)tok * EMBED;
  __bf16* dst = Emb + (long)row * EMBED;
  for (int c = threadIdx.x; c < EMBED; c += blockDim.x) dst[c] = (__bf16)src[c];
}

// EW = Emb @ W  (4080x1024 @ 1024x1024); 2 N-tiles per wave (A reused 2x)
__global__ __launch_bounds__(256) void k_ewgemm(const __bf16* __restrict__ Emb,
                                                const __bf16* __restrict__ Wb,
                                                float* __restrict__ EW) {
  __shared__ __bf16 sA[16 * EMBED];   // 32 KB row block
  int t = blockIdx.x;
  {
    const uint4* g = (const uint4*)(Emb + (long)t * 16 * EMBED);
    uint4* s = (uint4*)sA;
    for (int i = threadIdx.x; i < (16 * EMBED * 2) / 16; i += blockDim.x) s[i] = g[i];
  }
  __syncthreads();
  int wave = threadIdx.x >> 5, lane = threadIdx.x & 31;
  int n0 = (blockIdx.y * 8 + wave) * 32, n1 = n0 + 16;
  v8f acc0 = {}, acc1 = {};
  for (int kk = 0; kk < EMBED; kk += 32) {
    if (kk + 32 < EMBED)
      __builtin_prefetch((const void*)(Wb + (long)(kk + 32 + lane) * EMBED + n0), 0, 3);
    v16bf a = load_a_frag(sA, EMBED, kk);
    v16bf b0 = load_b_frag(Wb, EMBED, kk, n0);
    v16bf b1 = load_b_frag(Wb, EMBED, kk, n1);
    acc0 = wmma_bf16(a, b0, acc0);
    acc1 = wmma_bf16(a, b1, acc1);
  }
  int rbase = (lane >> 4) * 8, c0 = n0 + (lane & 15), c1 = n1 + (lane & 15);
  float* out = EW + (long)t * 16 * EMBED;
#pragma unroll
  for (int r = 0; r < 8; ++r) {
    out[(long)(rbase + r) * EMBED + c0] = acc0[r];
    out[(long)(rbase + r) * EMBED + c1] = acc1[r];
  }
}

// sequential recurrence: h_t = tanh(EW_t + h_{t-1} @ W2 + b); store bf16 H.
// one workgroup of 32 waves; h lives in LDS; each wave owns two 16-col tiles.
__global__ __launch_bounds__(1024) void k_recur(const float* __restrict__ EW,
                                                const __bf16* __restrict__ W2b,
                                                const float* __restrict__ bias,
                                                const float* __restrict__ h0,
                                                __bf16* __restrict__ Hbf) {
  __shared__ __bf16 hbuf[16 * EMBED];   // 32 KB
  int tid = threadIdx.x, wave = tid >> 5, lane = tid & 31;
  for (int i = tid; i < 16 * EMBED; i += blockDim.x)
    hbuf[i] = (__bf16)h0[i & (EMBED - 1)];
  __syncthreads();
  int rbase = (lane >> 4) * 8;
  int n0 = wave * 32, n1 = n0 + 16;
  int c0 = n0 + (lane & 15), c1 = n1 + (lane & 15);
  for (int t = 0; t < STEPS; ++t) {
    const float* ew = EW + (long)t * 16 * EMBED;
    v8f acc0 = {}, acc1 = {};
#pragma unroll
    for (int r = 0; r < 8; ++r) {
      acc0[r] = ew[(rbase + r) * EMBED + c0] + bias[c0];
      acc1[r] = ew[(rbase + r) * EMBED + c1] + bias[c1];
    }
    for (int kk = 0; kk < EMBED; kk += 32) {
      v16bf a = load_a_frag(hbuf, EMBED, kk);
      v16bf b0 = load_b_frag(W2b, EMBED, kk, n0);
      v16bf b1 = load_b_frag(W2b, EMBED, kk, n1);
      acc0 = wmma_bf16(a, b0, acc0);
      acc1 = wmma_bf16(a, b1, acc1);
    }
    __syncthreads();   // all waves done reading h_{t-1}
    __bf16* hout = Hbf + (long)t * 16 * EMBED;
#pragma unroll
    for (int r = 0; r < 8; ++r) {
      __bf16 v0 = (__bf16)tanhf(acc0[r]);
      __bf16 v1 = (__bf16)tanhf(acc1[r]);
      hbuf[(rbase + r) * EMBED + c0] = v0;
      hbuf[(rbase + r) * EMBED + c1] = v1;
      hout[(long)(rbase + r) * EMBED + c0] = v0;
      hout[(long)(rbase + r) * EMBED + c1] = v1;
    }
    __syncthreads();   // h_t visible before next step
  }
}

// ---- path A (big workspace): single GEMM pass, store fp32 logits + stats ----
__global__ __launch_bounds__(512) void k_logits_store(const __bf16* __restrict__ Hbf,
                                                      const __bf16* __restrict__ Vb,
                                                      const int* __restrict__ data,
                                                      float* __restrict__ Lbuf,
                                                      float* __restrict__ g_max,
                                                      float* __restrict__ g_label) {
  __shared__ __bf16 sA[16 * EMBED];
  __shared__ int   s_maxkey[16];
  __shared__ float s_label[16];
  __shared__ int   s_lbl[16];
  int t = blockIdx.x;
  int tid = threadIdx.x, wave = tid >> 5, lane = tid & 31;
  {
    const uint4* g = (const uint4*)(Hbf + (long)t * 16 * EMBED);
    uint4* s = (uint4*)sA;
    for (int i = tid; i < (16 * EMBED * 2) / 16; i += blockDim.x) s[i] = g[i];
  }
  if (tid < 16) {
    s_maxkey[tid] = (int)0x80000000;
    s_lbl[tid] = data[tid * TT + t + 1];
    s_label[tid] = 0.0f;
  }
  __syncthreads();
  int rbase = (lane >> 4) * 8, cl = lane & 15;
  float* Lrow = Lbuf + (long)t * 16 * HID;

  float lmax[8];
#pragma unroll
  for (int r = 0; r < 8; ++r) lmax[r] = -3.4e38f;

  for (int grp = wave; grp < NGRP; grp += 16) {
    int n0 = grp * 64;
    v8f acc[4] = {{}, {}, {}, {}};
    gemm_grp4(sA, Vb, n0, acc);
#pragma unroll
    for (int q = 0; q < 4; ++q) {
      int col = n0 + 16 * q + cl;
#pragma unroll
      for (int r = 0; r < 8; ++r) {
        float v = acc[q][r];
        int row = rbase + r;
        Lrow[(long)row * HID + col] = v;
        lmax[r] = fmaxf(lmax[r], v);
        if (col == s_lbl[row]) s_label[row] = v;  // unique writer
      }
    }
  }
#pragma unroll
  for (int r = 0; r < 8; ++r) {
    float m = lmax[r];
    for (int d = 1; d < 16; d <<= 1) m = fmaxf(m, __shfl_xor(m, d, 32));
    if (cl == 0) atomicMax(&s_maxkey[rbase + r], fkey(m));
  }
  __syncthreads();
  if (tid < 16) {
    g_max[t * 16 + tid] = funkey(s_maxkey[tid]);
    g_label[t * 16 + tid] = s_label[tid];
  }
}

// streaming stats over stored logits (memory-bound: ~261 MB @ 23.3 TB/s)
__global__ __launch_bounds__(512) void k_stats(const float* __restrict__ Lbuf,
                                               const float* __restrict__ g_max,
                                               const float* __restrict__ g_label,
                                               const int* __restrict__ topk_p,
                                               float* __restrict__ out) {
  __shared__ float s_sum[16];
  __shared__ int   s_cnt[16];
  int t = blockIdx.x, tid = threadIdx.x;
  if (tid < 16) { s_sum[tid] = 0.0f; s_cnt[tid] = 0; }
  __syncthreads();
  for (int row = 0; row < 16; ++row) {
    const float* L = Lbuf + ((long)t * 16 + row) * HID;
    float m = g_max[t * 16 + row], lv = g_label[t * 16 + row];
    float ls = 0.0f; int lc = 0;
    for (int c = tid; c < HID; c += blockDim.x) {
      float v = L[c];
      ls += __expf(v - m);
      lc += (v > lv) ? 1 : 0;
    }
    // wave-level reduce, then LDS atomics
    for (int d = 1; d < 32; d <<= 1) {
      ls += __shfl_xor(ls, d, 32);
      lc += __shfl_xor(lc, d, 32);
    }
    if ((tid & 31) == 0) { atomicAdd(&s_sum[row], ls); atomicAdd(&s_cnt[row], lc); }
  }
  __syncthreads();
  if (tid < 16) {
    float loss = g_max[t * 16 + tid] + logf(s_sum[tid]) - g_label[t * 16 + tid];
    atomicAdd(&out[0], loss);
    int topk = *topk_p;
    float corr = (s_cnt[tid] < topk) ? 1.0f : 0.0f;
    atomicAdd(&out[1], corr * (1.0f / (float)ROWS));
  }
}

// ---- path B (small workspace): fused two-pass, recompute from L2-resident V ----
__global__ __launch_bounds__(512) void k_logits_fused(const __bf16* __restrict__ Hbf,
                                                      const __bf16* __restrict__ Vb,
                                                      const int* __restrict__ data,
                                                      const int* __restrict__ topk_p,
                                                      float* __restrict__ out) {
  __shared__ __bf16 sA[16 * EMBED];
  __shared__ int   s_maxkey[16];
  __shared__ float s_label[16];
  __shared__ float s_sum[16];
  __shared__ int   s_cnt[16];
  __shared__ int   s_lbl[16];
  int t = blockIdx.x;
  int tid = threadIdx.x, wave = tid >> 5, lane = tid & 31;
  {
    const uint4* g = (const uint4*)(Hbf + (long)t * 16 * EMBED);
    uint4* s = (uint4*)sA;
    for (int i = tid; i < (16 * EMBED * 2) / 16; i += blockDim.x) s[i] = g[i];
  }
  if (tid < 16) {
    s_maxkey[tid] = (int)0x80000000;
    s_sum[tid] = 0.0f;
    s_cnt[tid] = 0;
    s_lbl[tid] = data[tid * TT + t + 1];
    s_label[tid] = 0.0f;
  }
  __syncthreads();
  int rbase = (lane >> 4) * 8, cl = lane & 15;

  // pass 1: row max + label logit
  float lmax[8];
#pragma unroll
  for (int r = 0; r < 8; ++r) lmax[r] = -3.4e38f;
  for (int grp = wave; grp < NGRP; grp += 16) {
    int n0 = grp * 64;
    v8f acc[4] = {{}, {}, {}, {}};
    gemm_grp4(sA, Vb, n0, acc);
#pragma unroll
    for (int q = 0; q < 4; ++q) {
      int col = n0 + 16 * q + cl;
#pragma unroll
      for (int r = 0; r < 8; ++r) {
        float v = acc[q][r];
        lmax[r] = fmaxf(lmax[r], v);
        if (col == s_lbl[rbase + r]) s_label[rbase + r] = v;
      }
    }
  }
#pragma unroll
  for (int r = 0; r < 8; ++r) {
    float m = lmax[r];
    for (int d = 1; d < 16; d <<= 1) m = fmaxf(m, __shfl_xor(m, d, 32));
    if (cl == 0) atomicMax(&s_maxkey[rbase + r], fkey(m));
  }
  __syncthreads();

  float rm[8], lv[8];
#pragma unroll
  for (int r = 0; r < 8; ++r) {
    rm[r] = funkey(s_maxkey[rbase + r]);
    lv[r] = s_label[rbase + r];
  }

  // pass 2: sum exp + count greater
  float lsum[8]; int lcnt[8];
#pragma unroll
  for (int r = 0; r < 8; ++r) { lsum[r] = 0.0f; lcnt[r] = 0; }
  for (int grp = wave; grp < NGRP; grp += 16) {
    int n0 = grp * 64;
    v8f acc[4] = {{}, {}, {}, {}};
    gemm_grp4(sA, Vb, n0, acc);
#pragma unroll
    for (int q = 0; q < 4; ++q) {
#pragma unroll
      for (int r = 0; r < 8; ++r) {
        float v = acc[q][r];
        lsum[r] += __expf(v - rm[r]);
        lcnt[r] += (v > lv[r]) ? 1 : 0;
      }
    }
  }
#pragma unroll
  for (int r = 0; r < 8; ++r) {
    float s = lsum[r]; int c = lcnt[r];
    for (int d = 1; d < 16; d <<= 1) {
      s += __shfl_xor(s, d, 32);
      c += __shfl_xor(c, d, 32);
    }
    if (cl == 0) { atomicAdd(&s_sum[rbase + r], s); atomicAdd(&s_cnt[rbase + r], c); }
  }
  __syncthreads();

  if (tid < 16) {
    float m = funkey(s_maxkey[tid]);
    float loss = m + logf(s_sum[tid]) - s_label[tid];
    atomicAdd(&out[0], loss);
    int topk = *topk_p;
    float corr = (s_cnt[tid] < topk) ? 1.0f : 0.0f;
    atomicAdd(&out[1], corr * (1.0f / (float)ROWS));
  }
}

// ---------------- launcher ----------------
extern "C" void kernel_launch(void* const* d_in, const int* in_sizes, int n_in,
                              void* d_out, int out_size, void* d_ws, size_t ws_size,
                              hipStream_t stream) {
  const int*   data = (const int*)d_in[0];
  const float* X    = (const float*)d_in[1];
  const float* W    = (const float*)d_in[2];
  const float* W2   = (const float*)d_in[3];
  const float* bias = (const float*)d_in[4];
  const float* V    = (const float*)d_in[5];
  const float* h0   = (const float*)d_in[6];
  const int*   topk = (const int*)d_in[7];
  float* out = (float*)d_out;

  char* ws = (char*)d_ws;
  size_t off = 0;
  __bf16* Wb  = (__bf16*)(ws + off); off += (size_t)EMBED * EMBED * 2;   //  2 MB
  __bf16* W2b = (__bf16*)(ws + off); off += (size_t)EMBED * EMBED * 2;   //  2 MB
  __bf16* Vb  = (__bf16*)(ws + off); off += (size_t)EMBED * HID * 2;     // 32 MB
  __bf16* Emb = (__bf16*)(ws + off); off += (size_t)ROWS * EMBED * 2;    //  8 MB
  float*  EW  = (float*)(ws + off);  off += (size_t)ROWS * EMBED * 4;    // 17 MB
  __bf16* Hbf = (__bf16*)(ws + off); off += (size_t)ROWS * EMBED * 2;    //  8 MB
  float* g_max   = (float*)(ws + off); off += (size_t)ROWS * 4;
  float* g_label = (float*)(ws + off); off += (size_t)ROWS * 4;
  off = (off + 255) & ~(size_t)255;
  float* Lbuf = (float*)(ws + off);
  size_t off_with_logits = off + (size_t)ROWS * HID * 4;                 // +261 MB
  (void)in_sizes; (void)n_in; (void)out_size;

  k_zero<<<1, 32, 0, stream>>>(out);
  k_convert<<<2048, 256, 0, stream>>>(W,  Wb,  EMBED * EMBED);
  k_convert<<<2048, 256, 0, stream>>>(W2, W2b, EMBED * EMBED);
  k_convert<<<2048, 256, 0, stream>>>(V,  Vb,  EMBED * HID);
  k_embed<<<ROWS, 256, 0, stream>>>(data, X, Emb);
  k_ewgemm<<<dim3(STEPS, 4), 256, 0, stream>>>(Emb, Wb, EW);
  k_recur<<<1, 1024, 0, stream>>>(EW, W2b, bias, h0, Hbf);

  if (ws_size >= off_with_logits) {
    // single GEMM pass + memory-bound streaming stats (saves ~134 GFLOP recompute)
    k_logits_store<<<STEPS, 512, 0, stream>>>(Hbf, Vb, data, Lbuf, g_max, g_label);
    k_stats<<<STEPS, 512, 0, stream>>>(Lbuf, g_max, g_label, topk, out);
  } else {
    // fused two-pass (V stays L2-resident; recompute instead of spilling)
    k_logits_fused<<<STEPS, 512, 0, stream>>>(Hbf, Vb, data, topk, out);
  }
}
